// EQGATEdgeGNN_77369540870666
// MI455X (gfx1250) — matrix operand order repaired and checked by reading, not compiled, not run on hardware
//
#include <hip/hip_runtime.h>
#include <hip/hip_bf16.h>

#define SDIM  256
#define VDIM  64
#define EDIM  32
#define NLAY  5
#define NNODE 4096
#define NEDGE 258048
#define DOUTC 417
#define DPAD  448
#define VROW  192   // 3*VDIM

typedef __bf16 bf16;
typedef __attribute__((ext_vector_type(8)))  bf16  v8bf;
typedef __attribute__((ext_vector_type(16))) bf16  v16bf;
typedef __attribute__((ext_vector_type(8)))  float v8f;
typedef int v4i __attribute__((vector_size(16)));   // matches builtin param type

// CDNA5 async global->LDS path (ASYNCcnt), guarded so a missing builtin
// falls back to direct L2 reads without breaking the compile.
#if defined(__has_builtin)
#if __has_builtin(__builtin_amdgcn_global_load_async_to_lds_b128) && \
    __has_builtin(__builtin_amdgcn_s_wait_asynccnt)
#define HAVE_ASYNC_LDS 1
#endif
#endif

#define AS1 __attribute__((address_space(1)))
#define AS3 __attribute__((address_space(3)))

__device__ __forceinline__ float fsilu(float x) {
    return x * (1.0f / (1.0f + __expf(-x)));
}

// A-fragment (16x32 bf16, M x K) per ISA 7.12.2: lanes 0-15 -> M=lane, K runs {0..7,16..23};
// lanes 16-31 -> M=lane-16, K runs {8..15,24..31}. Two 16B loads per lane.
__device__ __forceinline__ v16bf load_a_frag(const bf16* base, int ldm, int row0, int k0, int lane) {
    int half = lane >> 4;
    int m    = lane & 15;
    const bf16* r = base + (long)(row0 + m) * ldm + k0;
    v8bf lo = *(const v8bf*)(r + (half ? 8  : 0));
    v8bf hi = *(const v8bf*)(r + (half ? 24 : 16));
    return __builtin_shufflevector(lo, hi, 0,1,2,3,4,5,6,7,8,9,10,11,12,13,14,15);
}

// B-fragment from pre-packed buffer: frag = 32 lanes x 16 contiguous bf16 (one 32B load).
__device__ __forceinline__ v16bf load_b_frag(const bf16* P, int frag, int lane) {
    return *(const v16bf*)(P + ((long)frag << 9) + (lane << 4));
}

__device__ __forceinline__ v8f wmma_bf(v16bf a, v16bf b, v8f c) {
    return __builtin_amdgcn_wmma_f32_16x16x32_bf16(false, a, false, b, (short)0, c, false, false);
}

// ---------------- utility kernels ----------------

__global__ void copy_f32_kernel(float* dst, const float* src, long n) {
    for (long i = (long)blockIdx.x * blockDim.x + threadIdx.x; i < n;
         i += (long)gridDim.x * blockDim.x)
        dst[i] = src[i];
}

__global__ void f32_to_bf16_kernel(bf16* dst, const float* src, long n) {
    for (long i = (long)blockIdx.x * blockDim.x + threadIdx.x; i < n;
         i += (long)gridDim.x * blockDim.x)
        dst[i] = (bf16)src[i];
}

// Repack row-major fp32 W[K x NcSrc] into WMMA B-fragment bf16 layout, cols padded to NcPad.
__global__ void pack_b_kernel(const float* W, bf16* P, int K, int NcSrc, int NcPad) {
    int  KT = K >> 5, NT = NcPad >> 4;
    long total = (long)KT * NT * 512;
    for (long idx = (long)blockIdx.x * blockDim.x + threadIdx.x; idx < total;
         idx += (long)gridDim.x * blockDim.x) {
        int  j    = (int)(idx & 15);
        int  lane = (int)((idx >> 4) & 31);
        long frag = idx >> 9;
        int  nt   = (int)(frag % NT);
        int  kt   = (int)(frag / NT);
        int  n    = (nt << 4) + (lane & 15);
        int  k    = (kt << 5) + ((j >> 3) << 4) + ((lane >> 4) << 3) + (j & 7);
        float v   = (n < NcSrc) ? W[(long)k * NcSrc + n] : 0.0f;
        P[idx] = (bf16)v;
    }
}

// ---------------- per-layer node kernels ----------------

// LayerNorm s, RMS-norm v, zero accumulators; one block per node.
__global__ void __launch_bounds__(256)
norm_kernel(float* s, float* v, const float* gamma_i, const float* beta_i,
            bf16* s_bf, float* s_acc, float* v_acc, float* p_acc) {
    __shared__ float red[256];
    int  n = blockIdx.x, t = threadIdx.x;
    long sbase = (long)n * SDIM;
    float sv = s[sbase + t];

    red[t] = sv; __syncthreads();
    for (int off = 128; off > 0; off >>= 1) {
        if (t < off) red[t] += red[t + off];
        __syncthreads();
    }
    float mu = red[0] * (1.0f / SDIM); __syncthreads();
    float dv = sv - mu;
    red[t] = dv * dv; __syncthreads();
    for (int off = 128; off > 0; off >>= 1) {
        if (t < off) red[t] += red[t + off];
        __syncthreads();
    }
    float var = red[0] * (1.0f / SDIM);
    float sn = gamma_i[t] * dv * rsqrtf(var + 1e-6f) + beta_i[t];
    s[sbase + t]     = sn;
    s_bf[sbase + t]  = (bf16)sn;
    s_acc[sbase + t] = 0.0f;
    __syncthreads();

    long vbase = (long)n * VROW;
    float vv = (t < VROW) ? v[vbase + t] : 0.0f;
    red[t] = vv * vv; __syncthreads();
    for (int off = 128; off > 0; off >>= 1) {
        if (t < off) red[t] += red[t + off];
        __syncthreads();
    }
    float scale = rsqrtf(red[0] * (1.0f / VROW) + 1e-6f);
    if (t < VROW) {
        v[vbase + t]     = vv * scale;
        v_acc[vbase + t] = 0.0f;
    }
    if (t < 3) p_acc[(long)n * 3 + t] = 0.0f;
}

// s += s_acc/63, v += v_acc/63, p += p_acc/63; refresh bf16 copy of s.
__global__ void __launch_bounds__(256)
update_kernel(float* s, float* v, float* p,
              const float* s_acc, const float* v_acc, const float* p_acc, bf16* s_bf2) {
    int  n = blockIdx.x, t = threadIdx.x;
    const float inv = 1.0f / 63.0f;
    long sbase = (long)n * SDIM;
    float sv = s[sbase + t] + s_acc[sbase + t] * inv;
    s[sbase + t]     = sv;
    s_bf2[sbase + t] = (bf16)sv;
    if (t < VROW) {
        long vb = (long)n * VROW + t;
        v[vb] += v_acc[vb] * inv;
    }
    if (t < 3) {
        long pb = (long)n * 3 + t;
        p[pb] += p_acc[pb] * inv;
    }
}

// Subtract per-molecule centroid (64 atoms / molecule).
__global__ void __launch_bounds__(64)
center_kernel(float* p) {
    __shared__ float sx[64], sy[64], sz[64];
    int  t = threadIdx.x;
    long n = (long)blockIdx.x * 64 + t;
    float x = p[n * 3 + 0], y = p[n * 3 + 1], z = p[n * 3 + 2];
    sx[t] = x; sy[t] = y; sz[t] = z; __syncthreads();
    for (int off = 32; off > 0; off >>= 1) {
        if (t < off) { sx[t] += sx[t + off]; sy[t] += sy[t + off]; sz[t] += sz[t + off]; }
        __syncthreads();
    }
    const float inv = 1.0f / 64.0f;
    p[n * 3 + 0] = x - sx[0] * inv;
    p[n * 3 + 1] = y - sy[0] * inv;
    p[n * 3 + 2] = z - sz[0] * inv;
}

// Recompute per-edge geometry.
__global__ void geom_kernel(const float* p, const int* src, const int* dst,
                            float* dcur, float* adcur, float* rn) {
    long e = (long)blockIdx.x * blockDim.x + threadIdx.x;
    if (e >= NEDGE) return;
    int si = src[e], di = dst[e];
    float sx = p[(long)si * 3], sy = p[(long)si * 3 + 1], sz = p[(long)si * 3 + 2];
    float dx = p[(long)di * 3], dy = p[(long)di * 3 + 1], dz = p[(long)di * 3 + 2];
    float rx = dx - sx, ry = dy - sy, rz = dz - sz;
    float ss = fmaxf(rx * rx + ry * ry + rz * rz, 1e-6f);
    float dd = sqrtf(ss);
    float iv = 1.0f / dd;
    dcur[e]  = dd;
    adcur[e] = dx * sx + dy * sy + dz * sz;
    rn[e * 3 + 0] = rx * iv;
    rn[e * 3 + 1] = ry * iv;
    rn[e * 3 + 2] = rz * iv;
}

// ---------------- WMMA GEMM kernels ----------------

// C[M x 256] = op(A_bf16[M x K] @ Bpacked + bias). 4 waves/block, 4 N-tiles/wave:
// one A-fragment feeds 4 WMMAs (register-blocked). mode 0: f32 store; 1: silu->bf16; 2: f32 +=.
__global__ void __launch_bounds__(128)
gemm_node_kernel(const bf16* Abf, const bf16* Bp, const float* bias,
                 float* Cf, bf16* Cbf, float* Cacc, int K, int Nc, int mode) {
    int  lane = threadIdx.x & 31;
    int  wid  = threadIdx.x >> 5;
    int  NT   = Nc >> 4;
    int  nt0  = wid << 2;           // 4 n-tiles per wave
    long mt   = blockIdx.x;
    int  KT   = K >> 5;
    v8f acc[4] = {};
#pragma unroll 2
    for (int kt = 0; kt < KT; ++kt) {
        if (kt + 1 < KT)
            __builtin_prefetch(Bp + (((long)(kt + 1) * NT + nt0) << 9) + (lane << 4), 0, 1);
        v16bf a = load_a_frag(Abf, K, (int)(mt << 4), kt << 5, lane);
        acc[0] = wmma_bf(a, load_b_frag(Bp, kt * NT + nt0 + 0, lane), acc[0]);
        acc[1] = wmma_bf(a, load_b_frag(Bp, kt * NT + nt0 + 1, lane), acc[1]);
        acc[2] = wmma_bf(a, load_b_frag(Bp, kt * NT + nt0 + 2, lane), acc[2]);
        acc[3] = wmma_bf(a, load_b_frag(Bp, kt * NT + nt0 + 3, lane), acc[3]);
    }
#pragma unroll
    for (int j = 0; j < 4; ++j) {
        int col = ((nt0 + j) << 4) + (lane & 15);
#pragma unroll
        for (int k = 0; k < 8; ++k) {
            long row = (mt << 4) + k + ((lane >> 4) << 3);
            float val = acc[j][k];
            if (bias) val += bias[col];
            if (mode == 0)      Cf[row * Nc + col] = val;
            else if (mode == 1) Cbf[row * Nc + col] = (bf16)fsilu(val);
            else                Cacc[row * Nc + col] += val;
        }
    }
}

// ---------------- fused edge kernel ----------------
// Per 16-edge tile: h = silu(A[src]+B[dst]+e@We+feat2@Wda+b1); out = h@W2+b2;
// scatter into segment accumulators / e buffers. 4 waves/block, 16 edges/wave.
// We fragments staged block-wide into LDS via CDNA5 async global->LDS copies.
#define WPB 4
__global__ void __launch_bounds__(128)
edge_kernel(int layer,
            const float* A, const float* Bm, bf16* e_bf,
            const bf16* Wep, const bf16* W2p,
            const float* Wda_i, const float* b1_i, const float* b2_i,
            const float* v_n, const int* src, const int* dst,
            const float* dcur, const float* adcur, const float* rn,
            float* s_acc, float* v_acc, float* p_acc, float* e_out_f) {
    __shared__ bf16  hbuf[WPB][16 * SDIM];
    __shared__ int   s_srcl[WPB][16];
    __shared__ int   s_dstl[WPB][16];
    __shared__ float s_rn[WPB][16][3];
    __shared__ float s_dd[WPB][16];
    __shared__ float s_ad[WPB][16];
#ifdef HAVE_ASYNC_LDS
    __shared__ bf16  we_lds[EDIM * SDIM];   // 16 KB: all 16 We B-fragments
#endif

    int  lane = threadIdx.x & 31;
    int  wid  = threadIdx.x >> 5;
    long e0   = ((long)blockIdx.x * WPB + wid) * 16;

#ifdef HAVE_ASYNC_LDS
    {
        // 16 KB = 1024 x b128 transfers; 128 threads x 8 each.
        AS1 v4i* gsrc = (AS1 v4i*)(Wep);
        AS3 v4i* ldst = (AS3 v4i*)(we_lds);
#pragma unroll
        for (int it = 0; it < 8; ++it) {
            int idx = it * 128 + (int)threadIdx.x;
            __builtin_amdgcn_global_load_async_to_lds_b128(gsrc + idx, ldst + idx, 0, 0);
        }
    }
#endif

    if (lane < 16) {
        long eg = e0 + lane;
        s_srcl[wid][lane]  = src[eg];
        s_dstl[wid][lane]  = dst[eg];
        s_dd[wid][lane]    = dcur[eg];
        s_ad[wid][lane]    = adcur[eg];
        s_rn[wid][lane][0] = rn[eg * 3 + 0];
        s_rn[wid][lane][1] = rn[eg * 3 + 1];
        s_rn[wid][lane][2] = rn[eg * 3 + 2];
    }
#ifdef HAVE_ASYNC_LDS
    __builtin_amdgcn_s_wait_asynccnt(0);
    const bf16* WeUse = (const bf16*)we_lds;
#else
    const bf16* WeUse = Wep;
#endif
    __syncthreads();

    // Stage 1: h = silu(e@We + A[src] + B[dst] + feat2@Wda + b1), staged to LDS as bf16.
    v16bf ea = load_a_frag(e_bf + e0 * EDIM, EDIM, 0, 0, lane);
    for (int nt = 0; nt < SDIM / 16; ++nt) {
        v8f acc = {};
        acc = wmma_bf(ea, load_b_frag(WeUse, nt, lane), acc);
        int c = (nt << 4) + (lane & 15);
#pragma unroll
        for (int k = 0; k < 8; ++k) {
            int m  = k + ((lane >> 4) << 3);
            int sn = s_srcl[wid][m];
            int dn = s_dstl[wid][m];
            float t = acc[k]
                    + A[(long)sn * SDIM + c] + Bm[(long)dn * SDIM + c]
                    + s_dd[wid][m] * Wda_i[c] + s_ad[wid][m] * Wda_i[SDIM + c]
                    + b1_i[c];
            hbuf[wid][m * SDIM + c] = (bf16)fsilu(t);
        }
    }

    // Stage 2: out = h @ W2 (256 -> 448 padded). Hoist all 8 A-fragments of h into
    // registers once; inner loop is one 32B B-load per WMMA, streaming from L2.
    v16bf hfrag[8];
#pragma unroll
    for (int kt = 0; kt < 8; ++kt)
        hfrag[kt] = load_a_frag(&hbuf[wid][0], SDIM, 0, kt << 5, lane);

    for (int nt = 0; nt < DPAD / 16; ++nt) {
        v8f acc = {};
#pragma unroll
        for (int kt = 0; kt < 8; ++kt)
            acc = wmma_bf(hfrag[kt], load_b_frag(W2p, kt * (DPAD / 16) + nt, lane), acc);

        int col = (nt << 4) + (lane & 15);
#pragma unroll
        for (int k = 0; k < 8; ++k) {
            int  m  = k + ((lane >> 4) << 3);
            long eg = e0 + m;
            int  dn = s_dstl[wid][m];
            float val = acc[k];
            if (col < DOUTC) val += b2_i[col];
            if (col < SDIM) {
                atomicAdd(&s_acc[(long)dn * SDIM + col], val);
            } else if (col < SDIM + VDIM) {           // vg1: v[src] gate (layers > 0)
                if (layer > 0) {
                    int c  = col - SDIM;
                    int sn = s_srcl[wid][m];
#pragma unroll
                    for (int dd = 0; dd < 3; ++dd)
                        atomicAdd(&v_acc[(long)dn * VROW + dd * VDIM + c],
                                  v_n[(long)sn * VROW + dd * VDIM + c] * val);
                }
            } else if (col < SDIM + 2 * VDIM) {       // vg2: r_norm outer product
                int c = col - SDIM - VDIM;
#pragma unroll
                for (int dd = 0; dd < 3; ++dd)
                    atomicAdd(&v_acc[(long)dn * VROW + dd * VDIM + c],
                              s_rn[wid][m][dd] * val);
            } else if (col == SDIM + 2 * VDIM) {      // pg
#pragma unroll
                for (int dd = 0; dd < 3; ++dd)
                    atomicAdd(&p_acc[(long)dn * 3 + dd], s_rn[wid][m][dd] * val);
            } else if (col < DOUTC) {                 // new edge features
                int c = col - (SDIM + 2 * VDIM + 1);
                e_out_f[eg * EDIM + c] = val;
                e_bf[eg * EDIM + c]    = (bf16)val;
            }
        }
    }
}

// ---------------- host driver ----------------

extern "C" void kernel_launch(void* const* d_in, const int* in_sizes, int n_in,
                              void* d_out, int out_size, void* d_ws, size_t ws_size,
                              hipStream_t stream) {
    const float* s_in   = (const float*)d_in[0];
    const float* v_in   = (const float*)d_in[1];
    const float* p_in   = (const float*)d_in[2];
    const float* d_in_  = (const float*)d_in[3];
    const float* ad_in  = (const float*)d_in[4];
    const float* rn_in  = (const float*)d_in[5];
    const float* e_attr = (const float*)d_in[6];
    const int*   eig    = (const int*)d_in[7];
    const float* Ws  = (const float*)d_in[10];
    const float* Wt  = (const float*)d_in[11];
    const float* We  = (const float*)d_in[12];
    const float* Wda = (const float*)d_in[13];
    const float* b1  = (const float*)d_in[14];
    const float* W2  = (const float*)d_in[15];
    const float* b2  = (const float*)d_in[16];
    const float* Wu1 = (const float*)d_in[17];
    const float* bu1 = (const float*)d_in[18];
    const float* Wu2 = (const float*)d_in[19];
    const float* bu2 = (const float*)d_in[20];
    const float* gamma = (const float*)d_in[21];
    const float* beta  = (const float*)d_in[22];

    const int* src = eig;
    const int* dst = eig + NEDGE;

    // d_out layout: s | v | e | p
    float* s_out = (float*)d_out;
    float* v_out = s_out + (long)NNODE * SDIM;
    float* e_out = v_out + (long)NNODE * VROW;
    float* p_out = e_out + (long)NEDGE * EDIM;

    // workspace carve
    char* w = (char*)d_ws;
    auto carve = [&](size_t bytes) -> char* {
        char* r = w;
        w += (bytes + 255) & ~(size_t)255;
        return r;
    };
    float* A     = (float*)carve((size_t)NNODE * SDIM * 4);
    float* Bm    = (float*)carve((size_t)NNODE * SDIM * 4);
    float* s_acc = (float*)carve((size_t)NNODE * SDIM * 4);
    float* v_acc = (float*)carve((size_t)NNODE * VROW * 4);
    float* p_acc = (float*)carve((size_t)NNODE * 3 * 4);
    bf16*  s_nbf = (bf16*)carve((size_t)NNODE * SDIM * 2);
    bf16*  s_bf2 = (bf16*)carve((size_t)NNODE * SDIM * 2);
    bf16*  H1bf  = (bf16*)carve((size_t)NNODE * SDIM * 2);
    bf16*  e_bf  = (bf16*)carve((size_t)NEDGE * EDIM * 2);
    float* dcur  = (float*)carve((size_t)NEDGE * 4);
    float* adcur = (float*)carve((size_t)NEDGE * 4);
    float* rncur = (float*)carve((size_t)NEDGE * 3 * 4);
    bf16*  Wsp   = (bf16*)carve((size_t)NLAY * SDIM * SDIM * 2);
    bf16*  Wtp   = (bf16*)carve((size_t)NLAY * SDIM * SDIM * 2);
    bf16*  Wu1p  = (bf16*)carve((size_t)NLAY * SDIM * SDIM * 2);
    bf16*  Wu2p  = (bf16*)carve((size_t)NLAY * SDIM * SDIM * 2);
    bf16*  Wep   = (bf16*)carve((size_t)NLAY * EDIM * SDIM * 2);
    bf16*  W2p   = (bf16*)carve((size_t)NLAY * SDIM * DPAD * 2);

    // init state into d_out + mirrors
    copy_f32_kernel<<<1024, 256, 0, stream>>>(s_out, s_in, (long)NNODE * SDIM);
    copy_f32_kernel<<<1024, 256, 0, stream>>>(v_out, v_in, (long)NNODE * VROW);
    copy_f32_kernel<<<16,   256, 0, stream>>>(p_out, p_in, (long)NNODE * 3);
    copy_f32_kernel<<<2048, 256, 0, stream>>>(e_out, e_attr, (long)NEDGE * EDIM);
    copy_f32_kernel<<<1024, 256, 0, stream>>>(dcur, d_in_, (long)NEDGE);
    copy_f32_kernel<<<1024, 256, 0, stream>>>(adcur, ad_in, (long)NEDGE);
    copy_f32_kernel<<<2048, 256, 0, stream>>>(rncur, rn_in, (long)NEDGE * 3);
    f32_to_bf16_kernel<<<2048, 256, 0, stream>>>(e_bf, e_attr, (long)NEDGE * EDIM);

    // pack weights (bf16 B-fragment layout)
    for (int i = 0; i < NLAY; ++i) {
        pack_b_kernel<<<256, 256, 0, stream>>>(Ws  + (long)i * SDIM * SDIM, Wsp  + (long)i * SDIM * SDIM, SDIM, SDIM, SDIM);
        pack_b_kernel<<<256, 256, 0, stream>>>(Wt  + (long)i * SDIM * SDIM, Wtp  + (long)i * SDIM * SDIM, SDIM, SDIM, SDIM);
        pack_b_kernel<<<256, 256, 0, stream>>>(Wu1 + (long)i * SDIM * SDIM, Wu1p + (long)i * SDIM * SDIM, SDIM, SDIM, SDIM);
        pack_b_kernel<<<256, 256, 0, stream>>>(Wu2 + (long)i * SDIM * SDIM, Wu2p + (long)i * SDIM * SDIM, SDIM, SDIM, SDIM);
        pack_b_kernel<<<64,  256, 0, stream>>>(We  + (long)i * EDIM * SDIM, Wep  + (long)i * EDIM * SDIM, EDIM, SDIM, SDIM);
        pack_b_kernel<<<448, 256, 0, stream>>>(W2  + (long)i * SDIM * DOUTC, W2p + (long)i * SDIM * DPAD, SDIM, DOUTC, DPAD);
    }

    for (int i = 0; i < NLAY; ++i) {
        norm_kernel<<<NNODE, 256, 0, stream>>>(
            s_out, v_out, gamma + (long)i * SDIM, beta + (long)i * SDIM,
            s_nbf, s_acc, v_acc, p_acc);

        gemm_node_kernel<<<NNODE / 16, 128, 0, stream>>>(
            s_nbf, Wsp + (long)i * SDIM * SDIM, nullptr, A, nullptr, nullptr,
            SDIM, SDIM, 0);
        gemm_node_kernel<<<NNODE / 16, 128, 0, stream>>>(
            s_nbf, Wtp + (long)i * SDIM * SDIM, nullptr, Bm, nullptr, nullptr,
            SDIM, SDIM, 0);

        edge_kernel<<<NEDGE / (16 * WPB), 128, 0, stream>>>(
            i, A, Bm, e_bf,
            Wep + (long)i * EDIM * SDIM, W2p + (long)i * SDIM * DPAD,
            Wda + (long)i * 2 * SDIM, b1 + (long)i * SDIM, b2 + (long)i * DOUTC,
            v_out, src, dst, dcur, adcur, rncur,
            s_acc, v_acc, p_acc, e_out);

        update_kernel<<<NNODE, 256, 0, stream>>>(
            s_out, v_out, p_out, s_acc, v_acc, p_acc, s_bf2);

        if (i < NLAY - 1) {
            gemm_node_kernel<<<NNODE / 16, 128, 0, stream>>>(
                s_bf2, Wu1p + (long)i * SDIM * SDIM, bu1 + (long)i * SDIM,
                nullptr, H1bf, nullptr, SDIM, SDIM, 1);
            gemm_node_kernel<<<NNODE / 16, 128, 0, stream>>>(
                H1bf, Wu2p + (long)i * SDIM * SDIM, bu2 + (long)i * SDIM,
                nullptr, nullptr, s_out, SDIM, SDIM, 2);
        }

        center_kernel<<<64, 64, 0, stream>>>(p_out);
        geom_kernel<<<(NEDGE + 255) / 256, 256, 0, stream>>>(
            p_out, src, dst, dcur, adcur, rncur);
    }
}
